// Model_50070728737130
// MI455X (gfx1250) — matrix operand order repaired and checked by reading, not compiled, not run on hardware
//
#include <hip/hip_runtime.h>
#include <hip/hip_bf16.h>

#define B_SZ   32
#define N_PTS  1024
#define KNN    20
#define EPS_BN 1e-5f

typedef __attribute__((ext_vector_type(16))) _Float16 v16h;
typedef __attribute__((ext_vector_type(8)))  float    v8f;

// ---------------------------------------------------------------------------
// WMMA fragment helpers (CDNA5 16x16x32 f16, wave32 layouts per ISA 7.12.2)
// A (16xK row-major, stride ld halves): lane m=lane&15, h=lane>>4
//   halves[0..7]  = A[m][k0 + h*8 + 0..7]
//   halves[8..15] = A[m][k0 + 16 + h*8 + 0..7]
// B (element (k,n) at base[n*ld + k]): lane n=lane&15, h=lane>>4
//   halves[0..7]  = B[k0 + h*8 + 0..7][n]
//   halves[8..15] = B[k0 + 16 + h*8 + 0..7][n]
// C/D f32: VGPR r -> row = r + 8*h, col = lane&15
// ---------------------------------------------------------------------------
__device__ __forceinline__ v16h load_a_frag(const _Float16* __restrict__ base,
                                            int ld, int lane) {
  int m = lane & 15, h = lane >> 4;
  const _Float16* p = base + m * ld;
  v16h a;
#pragma unroll
  for (int t = 0; t < 8; ++t) {
    a[t]     = p[h * 8 + t];
    a[8 + t] = p[16 + h * 8 + t];
  }
  return a;
}

__device__ __forceinline__ v16h load_b_frag(const _Float16* __restrict__ base,
                                            int ld, int lane) {
  int n = lane & 15, h = lane >> 4;
  const _Float16* p = base + n * ld;
  v16h b;
#pragma unroll
  for (int t = 0; t < 8; ++t) {
    b[t]     = p[h * 8 + t];
    b[8 + t] = p[16 + h * 8 + t];
  }
  return b;
}

__device__ __forceinline__ v8f wmma_f16(v16h a, v16h b, v8f c) {
  return __builtin_amdgcn_wmma_f32_16x16x32_f16(false, a, false, b,
                                                (short)0, c, false, false);
}

// top-20 insertion (ascending distance, like top_k of -d)
__device__ __forceinline__ void topk_insert(float* bd, int* bi, float d, int i) {
  if (d < bd[KNN - 1]) {
    int t = KNN - 1;
    while (t > 0 && bd[t - 1] > d) {
      bd[t] = bd[t - 1];
      bi[t] = bi[t - 1];
      --t;
    }
    bd[t] = d;
    bi[t] = i;
  }
}

// ---------------------------------------------------------------------------
// Kernel 1: knn on pos (D=3).  grid = B*(N/256), block = 256
// ---------------------------------------------------------------------------
__global__ void knn_pos_kernel(const float* __restrict__ pos,
                               int* __restrict__ idx1) {
  __shared__ float sx[256 * 3];
  int b = blockIdx.x / (N_PTS / 256);
  int i = (blockIdx.x % (N_PTS / 256)) * 256 + threadIdx.x;
  const float* pb = pos + (size_t)b * N_PTS * 3;
  float xi0 = pb[i * 3 + 0], xi1 = pb[i * 3 + 1], xi2 = pb[i * 3 + 2];

  float bd[KNN];
  int   bi[KNN];
#pragma unroll
  for (int t = 0; t < KNN; ++t) { bd[t] = 3.4e38f; bi[t] = 0; }

  for (int m0 = 0; m0 < N_PTS; m0 += 256) {
    for (int t = threadIdx.x; t < 256 * 3; t += 256) sx[t] = pb[m0 * 3 + t];
    __syncthreads();
    for (int mm = 0; mm < 256; ++mm) {
      float d0 = xi0 - sx[mm * 3 + 0];
      float d1 = xi1 - sx[mm * 3 + 1];
      float d2 = xi2 - sx[mm * 3 + 2];
      topk_insert(bd, bi, d0 * d0 + d1 * d1 + d2 * d2, m0 + mm);
    }
    __syncthreads();
  }
  int* out = idx1 + (size_t)(b * N_PTS + i) * KNN;
#pragma unroll
  for (int t = 0; t < KNN; ++t) out[t] = bi[t];
}

// ---------------------------------------------------------------------------
// Kernel 2: conv1 edge MLP (6->64 BN ReLU ->64) + max over k.
// grid = B*N/4 blocks, block = 256 (4 points x 64 channels)
// ---------------------------------------------------------------------------
__global__ void conv1_kernel(const float* __restrict__ pos,
                             const int* __restrict__ idx1,
                             const float* __restrict__ w1,
                             const float* __restrict__ b1,
                             const float* __restrict__ g1,
                             const float* __restrict__ be1,
                             const float* __restrict__ m1,
                             const float* __restrict__ v1,
                             const float* __restrict__ w2,
                             const float* __restrict__ b2,
                             float* __restrict__ x1,
                             _Float16* __restrict__ x1h,
                             float* __restrict__ sq1) {
  __shared__ float sw1[6 * 64];
  __shared__ float sw2[64 * 64];
  __shared__ float sh1[4][64];
  int tid = threadIdx.x;
  for (int t = tid; t < 6 * 64; t += 256) sw1[t] = w1[t];
  for (int t = tid; t < 64 * 64; t += 256) sw2[t] = w2[t];
  __syncthreads();

  int lp = tid >> 6;   // local point 0..3
  int c  = tid & 63;   // channel
  int b  = blockIdx.x / (N_PTS / 4);
  int n  = (blockIdx.x % (N_PTS / 4)) * 4 + lp;
  int gi = b * N_PTS + n;

  float scc = g1[c] * rsqrtf(v1[c] + EPS_BN);
  float b1c = b1[c], m1c = m1[c], bec = be1[c], b2c = b2[c];

  float xi0 = pos[gi * 3 + 0], xi1 = pos[gi * 3 + 1], xi2 = pos[gi * 3 + 2];

  float acc = -3.4e38f;
  for (int kk = 0; kk < KNN; ++kk) {
    int j  = idx1[(size_t)gi * KNN + kk];
    int gj = b * N_PTS + j;
    float e0 = xi0, e1 = xi1, e2 = xi2;
    float e3 = pos[gj * 3 + 0] - xi0;
    float e4 = pos[gj * 3 + 1] - xi1;
    float e5 = pos[gj * 3 + 2] - xi2;
    float s = e0 * sw1[0 * 64 + c] + e1 * sw1[1 * 64 + c] + e2 * sw1[2 * 64 + c] +
              e3 * sw1[3 * 64 + c] + e4 * sw1[4 * 64 + c] + e5 * sw1[5 * 64 + c];
    float h = (s + b1c - m1c) * scc + bec;
    h = fmaxf(h, 0.0f);
    sh1[lp][c] = h;
    __syncthreads();
    float s2 = b2c;
#pragma unroll 8
    for (int j2 = 0; j2 < 64; ++j2) s2 += sh1[lp][j2] * sw2[j2 * 64 + c];
    acc = fmaxf(acc, s2);
    __syncthreads();
  }
  x1[(size_t)gi * 64 + c]  = acc;
  x1h[(size_t)gi * 64 + c] = (_Float16)acc;

  // squared norm per point (for knn2 distances)
  sh1[lp][c] = acc * acc;
  __syncthreads();
  if (c == 0) {
    float s = 0.0f;
    for (int j2 = 0; j2 < 64; ++j2) s += sh1[lp][j2];
    sq1[gi] = s;
  }
}

// ---------------------------------------------------------------------------
// Kernel 3: weight prep — transpose w3 / wl to N-major f16
// ---------------------------------------------------------------------------
__global__ void prep_weights_kernel(const float* __restrict__ w3,
                                    const float* __restrict__ wl,
                                    _Float16* __restrict__ w3T,
                                    _Float16* __restrict__ wlT) {
  int tid = blockIdx.x * 256 + threadIdx.x;
  if (tid < 128 * 128) {
    int n = tid / 128, k = tid % 128;
    w3T[n * 128 + k] = (_Float16)w3[k * 128 + n];  // w3:[128,128]
  }
  if (tid < 128 * 192) {
    int n = tid / 192, k = tid % 192;
    wlT[n * 192 + k] = (_Float16)wl[k * 128 + n];  // wl:[192,128]
  }
}

// ---------------------------------------------------------------------------
// Kernel 4: knn2 — WMMA gram tile (16 rows x 1024 cols, K=64) + top-20.
// grid = B*(N/16) blocks, block = 128 (4 waves). Two column passes, 32KB LDS.
// ---------------------------------------------------------------------------
__global__ void knn2_kernel(const _Float16* __restrict__ x1h,
                            const float* __restrict__ sq1,
                            int* __restrict__ idx2) {
  __shared__ float dist[16 * 512];  // 32 KB
  int b  = blockIdx.x >> 6;          // / (N/16)
  int i0 = (blockIdx.x & 63) << 4;
  const _Float16* X  = x1h + (size_t)b * N_PTS * 64;
  const float*    sq = sq1 + (size_t)b * N_PTS;
  int wave = threadIdx.x >> 5, lane = threadIdx.x & 31;

  v16h a0 = load_a_frag(X + i0 * 64, 64, lane);
  v16h a1 = load_a_frag(X + i0 * 64 + 32, 64, lane);

  float bd[KNN];
  int   bi[KNN];
#pragma unroll
  for (int t = 0; t < KNN; ++t) { bd[t] = 3.4e38f; bi[t] = 0; }

  for (int half = 0; half < 2; ++half) {
    for (int jt = wave; jt < 32; jt += 4) {
      int j0 = half * 512 + jt * 16;
      v16h b0 = load_b_frag(X + j0 * 64, 64, lane);      // B[k][n] = X[j0+n][k]
      v16h b1 = load_b_frag(X + j0 * 64 + 32, 64, lane);
      v8f c = {};
      c = wmma_f16(a0, b0, c);
      c = wmma_f16(a1, b1, c);
      int nn = lane & 15, h = lane >> 4;
      float sqj = sq[j0 + nn];
#pragma unroll
      for (int r = 0; r < 8; ++r) {
        int row = r + 8 * h;
        dist[row * 512 + jt * 16 + nn] = sq[i0 + row] + sqj - 2.0f * c[r];
      }
    }
    __syncthreads();
    if (threadIdx.x < 16) {
      int row = threadIdx.x;
      for (int m = 0; m < 512; ++m)
        topk_insert(bd, bi, dist[row * 512 + m], half * 512 + m);
    }
    __syncthreads();
  }
  if (threadIdx.x < 16) {
    int* out = idx2 + (size_t)(b * N_PTS + i0 + threadIdx.x) * KNN;
#pragma unroll
    for (int t = 0; t < KNN; ++t) out[t] = bi[t];
  }
}

// ---------------------------------------------------------------------------
// Kernel 5: conv2 — gather 80 edges (4 points), GEMM 80x128x128 via WMMA,
// max over k=20.  grid = B*N/4, block = 256 (8 waves, one N-tile each).
// LDS: E 20KB f16 + acc 40KB f32 + staging  (< 64KB)
// ---------------------------------------------------------------------------
__global__ void conv2_kernel(const float* __restrict__ x1,
                             const int* __restrict__ idx2,
                             const _Float16* __restrict__ w3T,
                             const float* __restrict__ b3,
                             _Float16* __restrict__ x2h) {
  __shared__ _Float16 E[80 * 128];
  __shared__ float    acc[80 * 128];
  __shared__ int      sidx[80];
  __shared__ float    sxi[4 * 64];
  int tid = threadIdx.x;
  int b   = blockIdx.x / (N_PTS / 4);
  int p0  = (blockIdx.x % (N_PTS / 4)) * 4;

  for (int t = tid; t < 80; t += 256)
    sidx[t] = idx2[(size_t)(b * N_PTS + p0 + t / KNN) * KNN + (t % KNN)];
  for (int t = tid; t < 4 * 64; t += 256)
    sxi[t] = x1[(size_t)(b * N_PTS + p0 + (t >> 6)) * 64 + (t & 63)];
  __syncthreads();

  // build edge features [x_i | x_j - x_i] in f16
  for (int t = tid; t < 80 * 128; t += 256) {
    int e = t >> 7, c = t & 127;
    int p = e / KNN;
    float v;
    if (c < 64) {
      v = sxi[p * 64 + c];
    } else {
      int j = sidx[e];
      v = x1[(size_t)(b * N_PTS + j) * 64 + (c - 64)] - sxi[p * 64 + (c - 64)];
    }
    E[t] = (_Float16)v;
  }
  __syncthreads();

  int wave = tid >> 5, lane = tid & 31;
  int n0 = wave * 16;
  v16h bf[4];
#pragma unroll
  for (int ks = 0; ks < 4; ++ks)
    bf[ks] = load_b_frag(w3T + n0 * 128 + ks * 32, 128, lane);

  int nn = lane & 15, h = lane >> 4;
  float bias = b3[n0 + nn];
#pragma unroll
  for (int mt = 0; mt < 5; ++mt) {
    v8f c = {};
#pragma unroll
    for (int ks = 0; ks < 4; ++ks) {
      v16h a = load_a_frag(E + (mt * 16) * 128 + ks * 32, 128, lane);
      c = wmma_f16(a, bf[ks], c);
    }
#pragma unroll
    for (int r = 0; r < 8; ++r)
      acc[(mt * 16 + r + 8 * h) * 128 + n0 + nn] = c[r] + bias;
  }
  __syncthreads();

  // max over k per (point, channel)
  for (int t = tid; t < 4 * 128; t += 256) {
    int p = t >> 7, cc = t & 127;
    float m = -3.4e38f;
#pragma unroll 4
    for (int kk = 0; kk < KNN; ++kk)
      m = fmaxf(m, acc[(p * KNN + kk) * 128 + cc]);
    x2h[(size_t)(b * N_PTS + p0 + p) * 128 + cc] = (_Float16)m;
  }
}

// ---------------------------------------------------------------------------
// Kernel 6: lin — hp = [x1|x2] @ wl + bl  (M=16 pts/block, N=128, K=192)
// grid = B*(N/16), block = 128 (4 waves x 2 N-tiles)
// ---------------------------------------------------------------------------
__global__ void lin_kernel(const _Float16* __restrict__ x1h,
                           const _Float16* __restrict__ x2h,
                           const _Float16* __restrict__ wlT,
                           const float* __restrict__ bl,
                           float* __restrict__ hp) {
  int b  = blockIdx.x >> 6;
  int i0 = (blockIdx.x & 63) << 4;
  int wave = threadIdx.x >> 5, lane = threadIdx.x & 31;

  const _Float16* X1 = x1h + (size_t)(b * N_PTS + i0) * 64;
  const _Float16* X2 = x2h + (size_t)(b * N_PTS + i0) * 128;
  v16h a[6];
  a[0] = load_a_frag(X1, 64, lane);
  a[1] = load_a_frag(X1 + 32, 64, lane);
#pragma unroll
  for (int s = 0; s < 4; ++s) a[2 + s] = load_a_frag(X2 + s * 32, 128, lane);

  int nn = lane & 15, h = lane >> 4;
#pragma unroll
  for (int q = 0; q < 2; ++q) {
    int n0 = (wave * 2 + q) * 16;
    v8f c = {};
#pragma unroll
    for (int ks = 0; ks < 6; ++ks) {
      v16h bf = load_b_frag(wlT + n0 * 192 + ks * 32, 192, lane);
      c = wmma_f16(a[ks], bf, c);
    }
    float bias = bl[n0 + nn];
#pragma unroll
    for (int r = 0; r < 8; ++r)
      hp[(size_t)(b * N_PTS + i0 + r + 8 * h) * 128 + n0 + nn] = c[r] + bias;
  }
}

// ---------------------------------------------------------------------------
// Kernel 7: global max over N -> out[B,128]
// ---------------------------------------------------------------------------
__global__ void maxpool_kernel(const float* __restrict__ hp,
                               float* __restrict__ out) {
  int b = blockIdx.x, c = threadIdx.x;
  float m = -3.4e38f;
  for (int n = 0; n < N_PTS; ++n)
    m = fmaxf(m, hp[(size_t)(b * N_PTS + n) * 128 + c]);
  out[b * 128 + c] = m;
}

// ---------------------------------------------------------------------------
extern "C" void kernel_launch(void* const* d_in, const int* in_sizes, int n_in,
                              void* d_out, int out_size, void* d_ws, size_t ws_size,
                              hipStream_t stream) {
  const float* pos = (const float*)d_in[0];
  const float* w1  = (const float*)d_in[1];
  const float* b1  = (const float*)d_in[2];
  const float* g1  = (const float*)d_in[3];
  const float* be1 = (const float*)d_in[4];
  const float* m1  = (const float*)d_in[5];
  const float* v1  = (const float*)d_in[6];
  const float* w2  = (const float*)d_in[7];
  const float* b2  = (const float*)d_in[8];
  const float* w3  = (const float*)d_in[9];
  const float* b3  = (const float*)d_in[10];
  const float* wl  = (const float*)d_in[11];
  const float* bl  = (const float*)d_in[12];
  float* out = (float*)d_out;

  char* ws = (char*)d_ws;
  size_t off = 0;
  auto carve = [&](size_t bytes) -> char* {
    char* p = ws + off;
    off = (off + bytes + 255) & ~(size_t)255;
    return p;
  };
  float*    x1   = (float*)   carve((size_t)B_SZ * N_PTS * 64 * 4);
  _Float16* x1h  = (_Float16*)carve((size_t)B_SZ * N_PTS * 64 * 2);
  float*    sq1  = (float*)   carve((size_t)B_SZ * N_PTS * 4);
  int*      idx1 = (int*)     carve((size_t)B_SZ * N_PTS * KNN * 4);
  int*      idx2 = (int*)     carve((size_t)B_SZ * N_PTS * KNN * 4);
  _Float16* x2h  = (_Float16*)carve((size_t)B_SZ * N_PTS * 128 * 2);
  _Float16* w3T  = (_Float16*)carve((size_t)128 * 128 * 2);
  _Float16* wlT  = (_Float16*)carve((size_t)128 * 192 * 2);
  float*    hp   = (float*)   carve((size_t)B_SZ * N_PTS * 128 * 4);
  (void)ws_size; (void)in_sizes; (void)n_in; (void)out_size;

  knn_pos_kernel<<<B_SZ * (N_PTS / 256), 256, 0, stream>>>(pos, idx1);

  conv1_kernel<<<B_SZ * N_PTS / 4, 256, 0, stream>>>(
      pos, idx1, w1, b1, g1, be1, m1, v1, w2, b2, x1, x1h, sq1);

  prep_weights_kernel<<<(128 * 192 + 255) / 256, 256, 0, stream>>>(w3, wl, w3T, wlT);

  knn2_kernel<<<B_SZ * (N_PTS / 16), 128, 0, stream>>>(x1h, sq1, idx2);

  conv2_kernel<<<B_SZ * N_PTS / 4, 256, 0, stream>>>(x1, idx2, w3T, b3, x2h);

  lin_kernel<<<B_SZ * (N_PTS / 16), 128, 0, stream>>>(x1h, x2h, wlT, bl, hp);

  maxpool_kernel<<<B_SZ, 128, 0, stream>>>(hp, out);
}
